// GCN_2353642078259
// MI455X (gfx1250) — matrix operand order repaired and checked by reading, not compiled, not run on hardware
//
#include <hip/hip_runtime.h>
#include <hip/hip_bf16.h>

// CDNA5 / gfx1250: wave32, WMMA 16x16x32 bf16 (fp32 accumulate).
typedef __attribute__((ext_vector_type(16))) __bf16 v16bf;
typedef __attribute__((ext_vector_type(2)))  __bf16 bf16x2;
typedef __attribute__((ext_vector_type(8)))  float  v8f;

// ---------------------------------------------------------------------------
// Degree / normalization kernels.  With self-loops deg >= 1, so
// dis = rsqrt(deg) exactly matches where(deg>0, rsqrt(max(deg,1)), 0).
// ---------------------------------------------------------------------------
__global__ void gcn_init_deg(float* __restrict__ deg, int n) {
    int i = blockIdx.x * blockDim.x + threadIdx.x;
    if (i < n) deg[i] = 1.0f;                       // self-loop contribution
}

__global__ void gcn_deg_accum(const int* __restrict__ ei, float* __restrict__ deg, int ne) {
    int e = blockIdx.x * blockDim.x + threadIdx.x;
    if (e < ne) atomicAdd(&deg[ei[(size_t)ne + e]], 1.0f);   // dst row of edge_index
}

__global__ void gcn_make_dis(float* __restrict__ deg, int n) {
    int i = blockIdx.x * blockDim.x + threadIdx.x;
    if (i < n) deg[i] = rsqrtf(deg[i]);             // in-place: deg -> dis
}

// ---------------------------------------------------------------------------
// WMMA fragment helpers (contiguous k = 16*(lane/16) + s, same in A and B, wave32).
// ---------------------------------------------------------------------------
// A-matrix 16x32 bf16: lane L<16 -> row L,  K = kbase+{0..15}
//                      lane L>=16 -> row L-16, K = kbase+{16..31}
__device__ __forceinline__ v16bf load_a_frag(const float* __restrict__ xrow,
                                             int kbase, int lane) {
    int koff = kbase + ((lane & 16) ? 16 : 0);
    const float4* p0 = (const float4*)(xrow + koff);        // K koff..koff+7
    const float4* p1 = (const float4*)(xrow + koff + 8);    // K koff+8..koff+15
    float4 f0 = p0[0], f1 = p0[1];
    float4 f2 = p1[0], f3 = p1[1];
    v16bf a;
    a[0] = (__bf16)f0.x; a[1] = (__bf16)f0.y; a[2]  = (__bf16)f0.z; a[3]  = (__bf16)f0.w;
    a[4] = (__bf16)f1.x; a[5] = (__bf16)f1.y; a[6]  = (__bf16)f1.z; a[7]  = (__bf16)f1.w;
    a[8] = (__bf16)f2.x; a[9] = (__bf16)f2.y; a[10] = (__bf16)f2.z; a[11] = (__bf16)f2.w;
    a[12]= (__bf16)f3.x; a[13]= (__bf16)f3.y; a[14] = (__bf16)f3.z; a[15] = (__bf16)f3.w;
    return a;
}

// B-matrix 32x16 bf16 from LDS.  LDS holds W pre-packed as K-pairs:
// sW[(k/2)*COUT + col] = {W[k][col], W[k+1][col]}.  Lane n<16 -> col n,
// K = kbase+0..15; lane n>=16 -> col n-16, K = kbase+16..31.  One 32-bit
// ds_load per VGPR, consecutive cols -> conflict-free banks (compiler pairs
// them into ds_load_2addr_stride64_b32).
template <int COUT>
__device__ __forceinline__ v16bf load_b_frag(const bf16x2* __restrict__ sW,
                                             int kbase, int colbase, int lane) {
    int col  = colbase + (lane & 15);
    int krow = kbase + ((lane & 16) ? 16 : 0);
    v16bf b;
#pragma unroll
    for (int v = 0; v < 8; ++v) {
        bf16x2 p = sW[((krow >> 1) + v) * COUT + col];
        b[2 * v]     = p.x;
        b[2 * v + 1] = p.y;
    }
    return b;
}

// ---------------------------------------------------------------------------
// H = X[nrows x CIN] @ W[CIN x COUT]   (bf16 WMMA, fp32 accumulate)
// Block = 256 threads = 8 waves; each wave owns 16 rows; block covers 128 rows.
// W converted to bf16 and staged in LDS once per block (CIN*COUT*2 bytes).
// Full blocks (all but the last) take an unguarded store path behind one
// block-uniform scalar branch; only the tail block pays per-element guards.
// ---------------------------------------------------------------------------
template <int CIN, int COUT>
__global__ void __launch_bounds__(256)
gcn_gemm_wmma(const float* __restrict__ X, const float* __restrict__ W,
              float* __restrict__ H, int nrows) {
    __shared__ bf16x2 sW[(CIN / 2) * COUT];

    int tid = threadIdx.x;
    for (int i = tid; i < (CIN / 2) * COUT; i += 256) {
        int kk = (i / COUT) * 2;
        int c  = i % COUT;
        bf16x2 p;
        p.x = (__bf16)W[(size_t)kk * COUT + c];
        p.y = (__bf16)W[(size_t)(kk + 1) * COUT + c];
        sW[i] = p;
    }
    __syncthreads();

    const int  wave      = tid >> 5;
    const int  lane      = tid & 31;
    const int  rowbase   = blockIdx.x * 128 + wave * 16;
    const bool fullblock = ((blockIdx.x + 1) * 128 <= nrows);   // block-uniform

    int arow = rowbase + (lane & 15);
    if (arow >= nrows) arow = nrows - 1;            // clamp: keep EXEC all-ones
    const float* xrow = X + (size_t)arow * CIN;

    v16bf afr[CIN / 32];
#pragma unroll
    for (int k = 0; k < CIN / 32; ++k) afr[k] = load_a_frag(xrow, k * 32, lane);

#pragma unroll
    for (int nt = 0; nt < COUT / 16; ++nt) {
        v8f c = {};
#pragma unroll
        for (int k = 0; k < CIN / 32; ++k) {
            v16bf b = load_b_frag<COUT>(sW, k * 32, nt * 16, lane);
            c = __builtin_amdgcn_wmma_f32_16x16x32_bf16(
                    false, afr[k], false, b, (short)0, c, false, false);
        }
        // C/D layout: VGPR v -> row rowbase + v (+8 for upper half-wave), col = lane&15
        int    col   = nt * 16 + (lane & 15);
        int    rbase = rowbase + ((lane & 16) ? 8 : 0);
        float* hp    = H + (size_t)rbase * COUT + col;
        if (fullblock) {
#pragma unroll
            for (int v = 0; v < 8; ++v) hp[(size_t)v * COUT] = c[v];
        } else {
#pragma unroll
            for (int v = 0; v < 8; ++v)
                if (rbase + v < nrows) hp[(size_t)v * COUT] = c[v];
        }
    }
}

// ---------------------------------------------------------------------------
// Edge scatter: one wave32 per edge; lane owns a contiguous float4/float2
// slice of the feature vector.  agg[dst] += H[src] * dis[src]*dis[dst].
// h/agg fit in the 192MB L2, so gather+atomic traffic stays on-die.
// ---------------------------------------------------------------------------
template <int C>
__global__ void __launch_bounds__(256)
gcn_scatter(const float* __restrict__ H, const int* __restrict__ ei,
            const float* __restrict__ dis, float* __restrict__ agg, int ne) {
    int wave = threadIdx.x >> 5;
    int lane = threadIdx.x & 31;
    int e    = blockIdx.x * 8 + wave;
    if (e >= ne) return;

    int s = ei[e];
    int d = ei[(size_t)ne + e];
    float w = dis[s] * dis[d];

    if constexpr (C == 128) {
        float4 hv = *(const float4*)(H + (size_t)s * C + lane * 4);
        float* od = agg + (size_t)d * C + lane * 4;
        atomicAdd(od + 0, hv.x * w);
        atomicAdd(od + 1, hv.y * w);
        atomicAdd(od + 2, hv.z * w);
        atomicAdd(od + 3, hv.w * w);
    } else {
        float2 hv = *(const float2*)(H + (size_t)s * C + lane * 2);
        float* od = agg + (size_t)d * C + lane * 2;
        atomicAdd(od + 0, hv.x * w);
        atomicAdd(od + 1, hv.y * w);
    }
}

// ---------------------------------------------------------------------------
// Fused epilogue: agg += H * dis^2 (self-loop, norm = dis[i]*dis[i]) + bias,
// optional ReLU.
// ---------------------------------------------------------------------------
template <int C, bool RELU>
__global__ void __launch_bounds__(256)
gcn_epilogue(float* __restrict__ agg, const float* __restrict__ H,
             const float* __restrict__ dis, const float* __restrict__ bias, int n) {
    long long idx   = (long long)blockIdx.x * blockDim.x + threadIdx.x;
    long long total = (long long)n * C;
    if (idx >= total) return;
    int node = (int)(idx / C);
    int c    = (int)(idx % C);
    float dd = dis[node];
    float v  = agg[idx] + H[idx] * dd * dd + bias[c];
    if (RELU) v = fmaxf(v, 0.0f);
    agg[idx] = v;
}

// ---------------------------------------------------------------------------
extern "C" void kernel_launch(void* const* d_in, const int* in_sizes, int n_in,
                              void* d_out, int out_size, void* d_ws, size_t ws_size,
                              hipStream_t stream) {
    const float* x  = (const float*)d_in[0];
    const int*   ei = (const int*)d_in[1];     // [2, E] int32: row0=src, row1=dst
    const float* W1 = (const float*)d_in[2];
    const float* b1 = (const float*)d_in[3];
    const float* W2 = (const float*)d_in[4];
    const float* b2 = (const float*)d_in[5];
    float* out = (float*)d_out;

    const int N = in_sizes[0] / 128;   // 100000
    const int E = in_sizes[1] / 2;     // 1600000

    // Workspace carve-out (256B aligned): dis[N], h1[N*128], a1[N*128], h2[N*64]
    auto al = [](size_t o) { return (o + 255) & ~(size_t)255; };
    char*  ws = (char*)d_ws;
    size_t o  = 0;
    float* dis = (float*)(ws + o); o = al(o + (size_t)N * 4);
    float* h1  = (float*)(ws + o); o = al(o + (size_t)N * 128 * 4);
    float* a1  = (float*)(ws + o); o = al(o + (size_t)N * 128 * 4);
    float* h2  = (float*)(ws + o); o = al(o + (size_t)N * 64 * 4);

    // Normalization coefficients
    gcn_init_deg <<<(N + 255) / 256, 256, 0, stream>>>(dis, N);
    gcn_deg_accum<<<(E + 255) / 256, 256, 0, stream>>>(ei, dis, E);
    gcn_make_dis <<<(N + 255) / 256, 256, 0, stream>>>(dis, N);

    // Layer 1: h1 = x @ W1 ; a1 = relu(segment_sum + selfloop + b1)
    gcn_gemm_wmma<128, 128><<<(N + 127) / 128, 256, 0, stream>>>(x, W1, h1, N);
    hipMemsetAsync(a1, 0, (size_t)N * 128 * 4, stream);
    gcn_scatter<128><<<(E + 7) / 8, 256, 0, stream>>>(h1, ei, dis, a1, E);
    {
        long long total = (long long)N * 128;
        gcn_epilogue<128, true><<<(unsigned)((total + 255) / 256), 256, 0, stream>>>(
            a1, h1, dis, b1, N);
    }

    // Layer 2: h2 = a1 @ W2 ; out = segment_sum + selfloop + b2
    gcn_gemm_wmma<128, 64><<<(N + 127) / 128, 256, 0, stream>>>(a1, W2, h2, N);
    hipMemsetAsync(out, 0, (size_t)N * 64 * 4, stream);
    gcn_scatter<64><<<(E + 7) / 8, 256, 0, stream>>>(h2, ei, dis, out, E);
    {
        long long total = (long long)N * 64;
        gcn_epilogue<64, false><<<(unsigned)((total + 255) / 256), 256, 0, stream>>>(
            out, h2, dis, b2, N);
    }
}
